// TextEncoder_4810363372568
// MI455X (gfx1250) — matrix operand order repaired and checked by reading, not compiled, not run on hardware
//
#include <hip/hip_runtime.h>
#include <hip/hip_bf16.h>

// ---------------------------------------------------------------------------
// Bidirectional GRU encoder for MI455X (gfx1250, wave32, WMMA, TDM).
//   V=32000, E=512, H=512, B=32, T=512
// Pipeline:
//   1) convert weights f32->bf16
//   2) embedding gather + bf16 convert
//   3) gx = x @ W_ih^T + b_ih  (bf16 WMMA; B-panel staged in LDS by the
//      Tensor Data Mover with pad_enable to avoid LDS bank conflicts)
//   4) persistent scan: 512 sequential steps, W_hh register-resident,
//      per-direction grid barrier (cluster-barrier fast path), gx prefetch.
// ---------------------------------------------------------------------------

typedef __attribute__((ext_vector_type(16))) __bf16 v16bf;
typedef __attribute__((ext_vector_type(8)))  float  v8f;
typedef unsigned int u32x4 __attribute__((ext_vector_type(4)));
typedef int          i32x8 __attribute__((ext_vector_type(8)));
typedef int          i32x4 __attribute__((ext_vector_type(4)));

#define B_  32
#define T_  512
#define E_  512
#define H_  512
#define G3  1536 // 3*H

union Frag {
    v16bf bf;
    uint4 q[2];
};

// A-matrix fragment (16x32 bf16, MxK):
// lane (hf=lane>>4, m=lane&15): 16 contiguous K values at k0 + hf*16.
__device__ __forceinline__ Frag load_a_frag(const __hip_bfloat16* __restrict__ A,
                                            int lda, int m0, int k0, int lane) {
    Frag f;
    const int hf = lane >> 4, mr = lane & 15;
    const __hip_bfloat16* row = A + (size_t)(m0 + mr) * lda;
    f.q[0] = *reinterpret_cast<const uint4*>(row + k0 + hf * 16);
    f.q[1] = *reinterpret_cast<const uint4*>(row + k0 + hf * 16 + 8);
    return f;
}

// B-matrix fragment (32x16 bf16, KxN) from W stored row-major N x K:
// lane (hf, n): 16 contiguous K values at k0 + hf*16.
__device__ __forceinline__ Frag load_b_frag(const __hip_bfloat16* __restrict__ W,
                                            int ldb, int n0, int k0, int lane) {
    Frag f;
    const int hf = lane >> 4, nc = lane & 15;
    const __hip_bfloat16* row = W + (size_t)(n0 + nc) * ldb;
    f.q[0] = *reinterpret_cast<const uint4*>(row + k0 + hf * 16);
    f.q[1] = *reinterpret_cast<const uint4*>(row + k0 + hf * 16 + 8);
    return f;
}

// Same, but from an LDS panel with padded row stride of 1040 bytes
// (TDM pad: +16B per 1024B row => row stride % 64 banks == 4 => conflict-free).
#define BPANEL_STRIDE 1040
__device__ __forceinline__ Frag load_b_frag_lds(const char* base, int row0, int k0, int lane) {
    Frag f;
    const int hf = lane >> 4, nc = lane & 15;
    const char* row = base + (size_t)(row0 + nc) * BPANEL_STRIDE;
    f.q[0] = *reinterpret_cast<const uint4*>(row + (k0 + hf * 16) * 2);
    f.q[1] = *reinterpret_cast<const uint4*>(row + (k0 + hf * 16) * 2 + 16);
    return f;
}

__device__ __forceinline__ float sigmoidf_(float x) {
    return 1.0f / (1.0f + __expf(-x));
}

// ---------------------------------------------------------------------------
// Kernel 1: convert the four weight matrices (each 1536x512) f32 -> bf16
// ---------------------------------------------------------------------------
__global__ void convert_w_kernel(const float* __restrict__ a, const float* __restrict__ b,
                                 const float* __restrict__ c, const float* __restrict__ d,
                                 __hip_bfloat16* __restrict__ oa, __hip_bfloat16* __restrict__ ob,
                                 __hip_bfloat16* __restrict__ oc, __hip_bfloat16* __restrict__ od,
                                 int n) {
    int i = blockIdx.x * blockDim.x + threadIdx.x;
    if (i < n) {
        oa[i] = __float2bfloat16(a[i]);
        ob[i] = __float2bfloat16(b[i]);
        oc[i] = __float2bfloat16(c[i]);
        od[i] = __float2bfloat16(d[i]);
    }
}

// ---------------------------------------------------------------------------
// Kernel 2: embedding gather + f32->bf16
// ---------------------------------------------------------------------------
__global__ void embed_bf16_kernel(const long long* __restrict__ src,
                                  const float* __restrict__ emb,
                                  __hip_bfloat16* __restrict__ xbf) {
    const int row = blockIdx.x; // b*T + t
    const long long v = src[row];
    const float* e = emb + (size_t)v * E_;
    __hip_bfloat16* o = xbf + (size_t)row * E_;
    for (int c = threadIdx.x; c < E_; c += blockDim.x)
        o[c] = __float2bfloat16(e[c]);
}

// ---------------------------------------------------------------------------
// Kernel 3: gx = x @ W_ih^T + b_ih   (M=16384, N=1536, K=512), per direction.
// Block = 8 waves, tile 128(M) x 64(N); wave tile 32x32 = 2x2 WMMA.
// B panel (64 rows x 512 K, 64KB bf16) is DMA'd into LDS once per block by
// the Tensor Data Mover, 8-byte elements, padded 16B per 1024B row.
// ---------------------------------------------------------------------------
__global__ void __launch_bounds__(256, 2)
gx_gemm_kernel(const __hip_bfloat16* __restrict__ xbf,
               const __hip_bfloat16* __restrict__ wih,   // [2][1536][512] bf16
               const float* __restrict__ bih0,
               const float* __restrict__ bih1,
               float* __restrict__ gx) {                 // [2][16384][1536] f32
    __shared__ __align__(16) char Bsh[64 * BPANEL_STRIDE]; // 66,560 B

    const int dir = blockIdx.z;
    const __hip_bfloat16* W = wih + (size_t)dir * G3 * E_;
    const float* bias = dir ? bih1 : bih0;
    float* out = gx + (size_t)dir * (size_t)(B_ * T_) * G3;

    const int tid = threadIdx.x, lane = tid & 31, w = tid >> 5;
    const int waveM = w & 3, waveN = w >> 2;
    const int n0base = blockIdx.y * 64;
    const int m0 = blockIdx.x * 128 + waveM * 32;
    const int n0 = n0base + waveN * 32;

#if __has_builtin(__builtin_amdgcn_tensor_load_to_lds)
    if (w == 0) {
        // Tensor DMA descriptor (CDNA5 ISA ch.8): 1D copy of 8192 x 8B with
        // LDS padding of 4 DWORDs (16B) every 128 x 8B (1024B).
        const unsigned long long ga =
            (unsigned long long)(const void*)(W + (size_t)n0base * E_);
        const unsigned la = (unsigned)(unsigned long long)(void*)&Bsh[0];
        u32x4 g0;
        g0[0] = 1u;                                   // count=1, user descriptor
        g0[1] = la;                                   // lds_addr
        g0[2] = (unsigned)ga;                         // global_addr[31:0]
        g0[3] = (unsigned)(ga >> 32) | 0x80000000u;   // addr[56:32] | type=2<<30
        i32x8 g1;
        g1[0] = 0x07C30000;  // data_size=8B, pad_enable, pad_interval=128*8B, pad=4 DW
        g1[1] = 0x20000000;  // tensor_dim0 = 8192 (low 16 in bits 31:16)
        g1[2] = 0x00010000;  // tensor_dim0 hi=0, tensor_dim1 = 1
        g1[3] = 0x20000000;  // tile_dim0 = 8192 (bits 31:16)
        g1[4] = 0;           // tile_dim1 = 0 (unused), tile_dim2 = 0
        g1[5] = 8192;        // tensor_dim0_stride
        g1[6] = 0;
        g1[7] = 0;
        i32x4 gz;
        gz[0] = 0; gz[1] = 0; gz[2] = 0; gz[3] = 0;
        i32x8 gz8;
        gz8[0] = 0; gz8[1] = 0; gz8[2] = 0; gz8[3] = 0;
        gz8[4] = 0; gz8[5] = 0; gz8[6] = 0; gz8[7] = 0;
        // 6-arg toolchain form: (g0, g1, g2, g3, extra group, cache-policy)
        __builtin_amdgcn_tensor_load_to_lds(g0, g1, gz, gz, gz8, 0);
        __builtin_amdgcn_s_wait_tensorcnt(0);
    }
#else
    // Fallback: cooperative copy replicating the padded layout.
    {
        const char* gsrc = (const char*)(W + (size_t)n0base * E_);
        for (int c = tid; c < 4096; c += 256) { // 4096 x 16B chunks
            const int row = c >> 6, wi = (c & 63) * 16;
            *reinterpret_cast<uint4*>(Bsh + row * BPANEL_STRIDE + wi) =
                *reinterpret_cast<const uint4*>(gsrc + c * 16);
        }
    }
#endif
    __syncthreads();

    v8f acc00 = {}, acc01 = {}, acc10 = {}, acc11 = {};
#pragma unroll 4
    for (int k0 = 0; k0 < E_; k0 += 32) {
        Frag a0 = load_a_frag(xbf, E_, m0,      k0, lane);
        Frag a1 = load_a_frag(xbf, E_, m0 + 16, k0, lane);
        Frag b0 = load_b_frag_lds(Bsh, waveN * 32,      k0, lane);
        Frag b1 = load_b_frag_lds(Bsh, waveN * 32 + 16, k0, lane);
        acc00 = __builtin_amdgcn_wmma_f32_16x16x32_bf16(false, a0.bf, false, b0.bf, (short)0, acc00, false, false);
        acc01 = __builtin_amdgcn_wmma_f32_16x16x32_bf16(false, a0.bf, false, b1.bf, (short)0, acc01, false, false);
        acc10 = __builtin_amdgcn_wmma_f32_16x16x32_bf16(false, a1.bf, false, b0.bf, (short)0, acc10, false, false);
        acc11 = __builtin_amdgcn_wmma_f32_16x16x32_bf16(false, a1.bf, false, b1.bf, (short)0, acc11, false, false);
    }

    const int hf = lane >> 4, lc = lane & 15;
    // C layout: VGPR v -> row m0 + v + 8*hf, col n0 + lc. gx is 192MB and
    // read-once much later: stream it past L2 with non-temporal stores.
#pragma unroll
    for (int v = 0; v < 8; ++v) {
        const int r0 = m0 + v + 8 * hf;
        const int r1 = r0 + 16;
        const int c0 = n0 + lc;
        const int c1 = n0 + 16 + lc;
        __builtin_nontemporal_store(acc00[v] + bias[c0], &out[(size_t)r0 * G3 + c0]);
        __builtin_nontemporal_store(acc01[v] + bias[c1], &out[(size_t)r0 * G3 + c1]);
        __builtin_nontemporal_store(acc10[v] + bias[c0], &out[(size_t)r1 * G3 + c0]);
        __builtin_nontemporal_store(acc11[v] + bias[c1], &out[(size_t)r1 * G3 + c1]);
    }
}

// ---------------------------------------------------------------------------
// Kernel 4: persistent recurrent scan. 16 WGs per direction, each owns a
// 32-wide hidden-column slice. W_hh fragments register-resident (128 VGPRs).
// Per-direction step barrier: hardware cluster barrier when dispatched in a
// 16-WG cluster, else per-direction atomic counter.
// ---------------------------------------------------------------------------
__global__ void __launch_bounds__(192, 1)
gru_scan_kernel(const __hip_bfloat16* __restrict__ whh,  // [2][1536][512] bf16
                const float* __restrict__ gx,            // [2][B][T][1536]
                const float* __restrict__ bhh0,
                const float* __restrict__ bhh1,
                const unsigned char* __restrict__ mask,  // [B][T] bool
                float* __restrict__ h32,                 // [2][B][H] f32
                __hip_bfloat16* __restrict__ hbf,        // [2 bufs][2][B][H] bf16
                float* __restrict__ out,                 // [B][T][2H] f32
                unsigned int* __restrict__ ctr) {        // 2 counters, 256B apart
    __shared__ float gh[32 * 100];

    const int blk  = blockIdx.x;
    const int dir  = blk >> 4;
    const int c0   = (blk & 15) * 32;
    const int tid  = threadIdx.x;
    const int lane = tid & 31;
    const int w    = tid >> 5;               // 0..5
    const int g    = w >> 1;                 // gate 0..2 (r,z,n)
    const int j0   = c0 + (w & 1) * 16;
    const int qb   = w * 16;

    const __hip_bfloat16* Whh = whh + (size_t)dir * G3 * H_;
    const float* gxd = gx + (size_t)dir * (size_t)(B_ * T_) * G3;
    const float* bhh = dir ? bhh1 : bhh0;
    float* h32d = h32 + (size_t)dir * B_ * H_;
    unsigned int* myctr = ctr + dir * 64;    // separate cachelines per direction
    const int cid = __builtin_amdgcn_cluster_id_x();

    // Hoist W_hh fragments: 16 K-chunks x 8 VGPRs = 128 VGPRs, loop-invariant.
    Frag bw[16];
#pragma unroll
    for (int kk = 0; kk < 16; ++kk)
        bw[kk] = load_b_frag(Whh, H_, g * H_ + j0, kk * 32, lane);

    const int hf = lane >> 4, lc = lane & 15;
    const size_t HBUF = (size_t)2 * B_ * H_;

    for (int s = 0; s < T_; ++s) {
        const int t = dir ? (T_ - 1 - s) : s;
        const __hip_bfloat16* hrd = hbf + (size_t)(s & 1) * HBUF + (size_t)dir * B_ * H_;
        __hip_bfloat16* hwr = hbf + (size_t)((s & 1) ^ 1) * HBUF + (size_t)dir * B_ * H_;

        // Prefetch this step's gx gate rows (HBM-resident) behind the GEMM.
        {
            const int pb = tid & 31, q = tid >> 5;          // 32 x 6 = 192
            const float* pf = gxd + (size_t)(pb * T_ + t) * G3
                              + (q >> 1) * H_ + c0 + (q & 1) * 16;
            __builtin_prefetch(pf, 0, 1);
        }

        // gh slice = h_prev(32x512) @ W_hh^T(96 slice cols)
        v8f acc0 = {}, acc1 = {};
#pragma unroll
        for (int kk = 0; kk < 16; ++kk) {
            Frag a0 = load_a_frag(hrd, H_, 0,  kk * 32, lane);
            Frag a1 = load_a_frag(hrd, H_, 16, kk * 32, lane);
            acc0 = __builtin_amdgcn_wmma_f32_16x16x32_bf16(false, a0.bf, false, bw[kk].bf, (short)0, acc0, false, false);
            acc1 = __builtin_amdgcn_wmma_f32_16x16x32_bf16(false, a1.bf, false, bw[kk].bf, (short)0, acc1, false, false);
        }
#pragma unroll
        for (int v = 0; v < 8; ++v) {
            gh[(v + 8 * hf) * 100 + qb + lc]      = acc0[v];
            gh[(16 + v + 8 * hf) * 100 + qb + lc] = acc1[v];
        }
        __syncthreads();

        // Elementwise gate math for this 32x32 hidden slice.
        for (int idx = tid; idx < B_ * 32; idx += 192) {
            const int b  = idx >> 5;
            const int jj = idx & 31;
            const int j  = c0 + jj;
            const float* gx_bt = gxd + (size_t)(b * T_ + t) * G3;
            const float xr = gx_bt[j];
            const float xz = gx_bt[H_ + j];
            const float xn = gx_bt[2 * H_ + j];
            const float hr = gh[b * 100 + jj]      + bhh[j];
            const float hz = gh[b * 100 + 32 + jj] + bhh[H_ + j];
            const float hn = gh[b * 100 + 64 + jj] + bhh[2 * H_ + j];
            const float hp = h32d[b * H_ + j];
            const float m  = mask[b * T_ + t] ? 1.0f : 0.0f;

            const float r = sigmoidf_(xr + hr);
            const float z = sigmoidf_(xz + hz);
            const float n = tanhf(xn + r * hn);
            const float hnew = (1.0f - z) * n + z * hp;
            const float h = m * hnew + (1.0f - m) * hp;

            __builtin_nontemporal_store(h, &out[(size_t)(b * T_ + t) * (2 * H_) + dir * H_ + j]);
            h32d[b * H_ + j] = h;
            hwr[b * H_ + j]  = __float2bfloat16(h);
        }

        // --- per-direction step barrier (16 WGs) ---
        __threadfence();          // release this thread's h writes (agent scope)
        __syncthreads();
        if (cid != 0) {
            // Hardware path: 16-WG cluster on one SE; wave 0 signals+waits the
            // cluster barrier (s_barrier_signal -3 / s_barrier_wait -3).
            if (w == 0) __builtin_amdgcn_s_cluster_barrier();
            __syncthreads();
        } else {
            if (tid == 0) {
                __hip_atomic_fetch_add(myctr, 1u, __ATOMIC_RELEASE, __HIP_MEMORY_SCOPE_AGENT);
                const unsigned int target = 16u * (unsigned int)(s + 1);
                while (__hip_atomic_load(myctr, __ATOMIC_ACQUIRE, __HIP_MEMORY_SCOPE_AGENT) < target)
                    __builtin_amdgcn_s_sleep(1);
            }
            __syncthreads();
        }
        __threadfence();          // acquire other WGs' h writes
    }
}

// ---------------------------------------------------------------------------
// Workspace layout (bytes)
// ---------------------------------------------------------------------------
static constexpr size_t OFF_XBF = 0;                                  // 16,777,216
static constexpr size_t OFF_WIH = 16777216;                           //  3,145,728
static constexpr size_t OFF_WHH = 19922944;                           //  3,145,728
static constexpr size_t OFF_GX  = 23068672;                           // 201,326,592
static constexpr size_t OFF_H32 = 224395264;                          //    131,072
static constexpr size_t OFF_HBF = 224526336;                          //    131,072 (2 bufs)
static constexpr size_t OFF_CTR = 224657408;                          //        512
static constexpr size_t ZERO_BYTES = 131072 + 131072 + 512;

extern "C" void kernel_launch(void* const* d_in, const int* in_sizes, int n_in,
                              void* d_out, int out_size, void* d_ws, size_t ws_size,
                              hipStream_t stream) {
    const long long*     src    = (const long long*)d_in[0];     // int64 [B,T]
    const unsigned char* mask   = (const unsigned char*)d_in[1]; // bool  [B,T]
    const float*         emb    = (const float*)d_in[2];
    const float*         W_ih_f = (const float*)d_in[3];
    const float*         W_hh_f = (const float*)d_in[4];
    const float*         b_ih_f = (const float*)d_in[5];
    const float*         b_hh_f = (const float*)d_in[6];
    const float*         W_ih_b = (const float*)d_in[7];
    const float*         W_hh_b = (const float*)d_in[8];
    const float*         b_ih_b = (const float*)d_in[9];
    const float*         b_hh_b = (const float*)d_in[10];

    char* ws = (char*)d_ws;
    __hip_bfloat16* xbf  = (__hip_bfloat16*)(ws + OFF_XBF);
    __hip_bfloat16* wih  = (__hip_bfloat16*)(ws + OFF_WIH); // [2][1536][512]
    __hip_bfloat16* whh  = (__hip_bfloat16*)(ws + OFF_WHH); // [2][1536][512]
    float*          gxb  = (float*)(ws + OFF_GX);           // [2][16384][1536]
    float*          h32  = (float*)(ws + OFF_H32);
    __hip_bfloat16* hbf  = (__hip_bfloat16*)(ws + OFF_HBF);
    unsigned int*   ctr  = (unsigned int*)(ws + OFF_CTR);

    // Zero hidden state + barrier counters (capture-safe).
    (void)hipMemsetAsync(ws + OFF_H32, 0, ZERO_BYTES, stream);

    // 1) weights -> bf16 (dir0: W_ih_f/W_hh_f, dir1: W_ih_b/W_hh_b)
    const int nW = G3 * E_; // 786432
    convert_w_kernel<<<(nW + 255) / 256, 256, 0, stream>>>(
        W_ih_f, W_hh_f, W_ih_b, W_hh_b,
        wih, whh, wih + (size_t)nW, whh + (size_t)nW, nW);

    // 2) embedding gather -> bf16
    embed_bf16_kernel<<<B_ * T_, 256, 0, stream>>>(src, emb, xbf);

    // 3) input projection GEMM, both directions
    dim3 ggrid(128, 24, 2); // M/128, N/64, dir
    gx_gemm_kernel<<<ggrid, 256, 0, stream>>>(xbf, wih, b_ih_f, b_ih_b, gxb);

    // 4) recurrent scan (persistent, 32 WGs = 16 per direction)
    gru_scan_kernel<<<32, 192, 0, stream>>>(whh, gxb, b_hh_f, b_hh_b, mask,
                                            h32, hbf, (float*)d_out, ctr);
}
